// AttnDecoderRNN_3822520894048
// MI455X (gfx1250) — matrix operand order repaired and checked by reading, not compiled, not run on hardware
//
#include <hip/hip_runtime.h>
#include <stdint.h>
#include <stddef.h>

// ---------------------------------------------------------------------------
// AttnDecoderRNN single step for MI455X (gfx1250, wave32, WMMA bf16 + TDM)
// B=16 H=32 W=128 V=512 HS=256 ENC=1024, P=H*W=4096
// et1 intermediate: bf16, NHWC with 1-pixel zero halo: [b][34][130][256]
// ---------------------------------------------------------------------------

typedef __attribute__((ext_vector_type(16))) __bf16          bf16x16;
typedef __attribute__((ext_vector_type(8)))  float           f32x8;
typedef __attribute__((ext_vector_type(8)))  unsigned short  u16x8;
typedef __attribute__((ext_vector_type(4)))  unsigned int    u32x4;
typedef __attribute__((ext_vector_type(8)))  int             i32x8;
typedef __attribute__((ext_vector_type(4)))  int             i32x4;

struct U16x16 { u16x8 lo, hi; };

#if defined(__has_builtin)
#if __has_builtin(__builtin_amdgcn_tensor_load_to_lds) && __has_builtin(__builtin_amdgcn_s_wait_tensorcnt)
#define HAVE_TDM 1
#endif
#endif

static __device__ inline unsigned short f2bf(float f) {
  union { float f; unsigned int u; } v; v.f = f;
  unsigned int r = v.u + 0x7FFFu + ((v.u >> 16) & 1u);   // RNE
  return (unsigned short)(r >> 16);
}

static __device__ inline bf16x16 mk_frag(const unsigned short* p_lo,
                                         const unsigned short* p_hi) {
  U16x16 u;
  u.lo = *reinterpret_cast<const u16x8*>(p_lo);   // ds_load_b128
  u.hi = *reinterpret_cast<const u16x8*>(p_hi);   // ds_load_b128
  return __builtin_bit_cast(bf16x16, u);
}

static __device__ inline float sigmoidf(float x) { return 1.f / (1.f + expf(-x)); }

#ifdef HAVE_TDM
// Issue one TDM load:  global (bf16, 2B elements) -> LDS, up to 3-D tile.
// LDS receives tile in [z][y][x] order, x contiguous.
// addr math: gaddr + 2*(x + y*stride0 + z*stride1)
static __device__ inline void tdm_load_bf16(
    const void* gptr, void* lptr,
    unsigned tensor_d0, unsigned tensor_d1, unsigned tensor_d2,
    unsigned tile_d0, unsigned tile_d1, unsigned tile_d2,
    unsigned stride0, unsigned stride1)
{
  const unsigned long long ga = (unsigned long long)(uintptr_t)gptr;
  const unsigned lds = (unsigned)(uintptr_t)lptr;   // low 32 bits = LDS offset
  u32x4 g0; i32x8 g1; i32x4 g2; i32x4 g3;
  g0[0] = 1u;                                          // count=1, no gather
  g0[1] = lds;                                         // lds_addr
  g0[2] = (unsigned)(ga & 0xFFFFFFFFu);                // global_addr lo
  g0[3] = (unsigned)((ga >> 32) & 0x01FFFFFFu) | (2u << 30);  // addr hi | type=2
  g1[0] = (int)(1u << 16);                             // wg_mask=0, data_size=2B
  g1[1] = (int)((tensor_d0 & 0xFFFFu) << 16);          // tensor_dim0 lo16
  g1[2] = (int)(((tensor_d0 >> 16) & 0xFFFFu) | ((tensor_d1 & 0xFFFFu) << 16));
  g1[3] = (int)(((tensor_d1 >> 16) & 0xFFFFu) | ((tile_d0 & 0xFFFFu) << 16));
  g1[4] = (int)((tile_d1 & 0xFFFFu) | ((tile_d2 & 0xFFFFu) << 16));
  g1[5] = (int)stride0;                                // tensor_dim0_stride lo32
  g1[6] = (int)((stride1 & 0xFFFFu) << 16);            // d0s hi16=0 | d1s lo16
  g1[7] = (int)((stride1 >> 16) & 0xFFFFFFFFu);        // d1s bits 47:16
  g2[0] = (int)tensor_d2; g2[1] = 0; g2[2] = 0; g2[3] = 0;
  g3[0] = 0; g3[1] = 0; g3[2] = 0; g3[3] = 0;
#if __clang_major__ >= 23
  i32x8 gx; gx[0]=0;gx[1]=0;gx[2]=0;gx[3]=0;gx[4]=0;gx[5]=0;gx[6]=0;gx[7]=0;
  __builtin_amdgcn_tensor_load_to_lds(g0, g1, g2, g3, gx, 0);
#else
  __builtin_amdgcn_tensor_load_to_lds(g0, g1, g2, g3, 0);
#endif
}
#endif

// ---------------------------------------------------------------------------
// K0: embedding gather + GRU1 + hidden1 projection.  grid=B, block=256.
// ---------------------------------------------------------------------------
__global__ void __launch_bounds__(256) k0_prep(
    const int* __restrict__ input_a, const float* __restrict__ hidden,
    const float* __restrict__ emb_table,
    const float* __restrict__ wih, const float* __restrict__ whh,
    const float* __restrict__ bih, const float* __restrict__ bhh,
    const float* __restrict__ hidden_w, const float* __restrict__ hidden_b,
    float* __restrict__ ws_emb, float* __restrict__ ws_st, float* __restrict__ ws_h1)
{
  __shared__ float lemb[256], lhid[256], lst[256];
  const int b = blockIdx.x, t = threadIdx.x;
  const int a = input_a[b];
  const float e = emb_table[a * 256 + t];
  lemb[t] = e; ws_emb[b * 256 + t] = e;
  lhid[t] = hidden[b * 256 + t];
  __syncthreads();
  float gr = 0.f, gz = 0.f, gn = 0.f, hr = 0.f, hz = 0.f, hn = 0.f;
  for (int k = 0; k < 256; ++k) {
    const float xe = lemb[k], xh = lhid[k];
    gr += xe * wih[(t)       * 256 + k];
    gz += xe * wih[(256 + t) * 256 + k];
    gn += xe * wih[(512 + t) * 256 + k];
    hr += xh * whh[(t)       * 256 + k];
    hz += xh * whh[(256 + t) * 256 + k];
    hn += xh * whh[(512 + t) * 256 + k];
  }
  const float r = sigmoidf(gr + bih[t]       + hr + bhh[t]);
  const float z = sigmoidf(gz + bih[256 + t] + hz + bhh[256 + t]);
  const float n = tanhf(gn + bih[512 + t] + r * (hn + bhh[512 + t]));
  const float st = (1.f - z) * n + z * lhid[t];
  lst[t] = st; ws_st[b * 256 + t] = st;
  __syncthreads();
  float acc = hidden_b[t];
  for (int k = 0; k < 256; ++k) acc += lst[k] * hidden_w[t * 256 + k];
  ws_h1[b * 256 + t] = acc;
}

// ---------------------------------------------------------------------------
// K1: attention_sum += conv3x3(decoder_attention) + bias.  grid=B*P/256.
// ---------------------------------------------------------------------------
__global__ void __launch_bounds__(256) k1_att(
    const float* __restrict__ att_in, const float* __restrict__ dec,
    const float* __restrict__ cw, const float* __restrict__ cb,
    float* __restrict__ att_out)
{
  const int i = blockIdx.x * 256 + threadIdx.x;      // < 65536
  const int b = i >> 12, p = i & 4095, h = p >> 7, w = p & 127;
  float acc = att_in[i] + cb[0];
  #pragma unroll
  for (int kh = 0; kh < 3; ++kh)
    #pragma unroll
    for (int kw = 0; kw < 3; ++kw) {
      const int hh = h + kh - 1, ww = w + kw - 1;
      if ((unsigned)hh < 32u && (unsigned)ww < 128u)
        acc += dec[(b << 12) + (hh << 7) + ww] * cw[kh * 3 + kw];
    }
  att_out[i] = acc;
}

// ---------------------------------------------------------------------------
// K_wprep: convtan_w [oc][ic][3][3] f32 -> bf16 [tap][oc][ic].  589824 elems.
// ---------------------------------------------------------------------------
__global__ void __launch_bounds__(256) k_wprep(
    const float* __restrict__ cw, unsigned short* __restrict__ cwbf)
{
  const int i = blockIdx.x * 256 + threadIdx.x;
  if (i < 9 * 256 * 256) {
    const int t = i >> 16, r = i & 65535, oc = r >> 8, ic = r & 255;
    cwbf[i] = f2bf(cw[((size_t)oc * 256 + ic) * 9 + t]);
  }
}

// ---------------------------------------------------------------------------
// K_zero: clear the 1-pixel halo of et1 [b][34][130][256].  1327104 elems.
// ---------------------------------------------------------------------------
__global__ void __launch_bounds__(256) k_zero_halo(unsigned short* __restrict__ et1)
{
  const int i = blockIdx.x * 256 + threadIdx.x;     // < 16*82944
  const int b = i / 82944;
  const int j = i - b * 82944;
  int row, col, ic;
  if (j < 66560) {                                   // top + bottom full rows
    row = (j < 33280) ? 0 : 33;
    const int jj = (j < 33280) ? j : j - 33280;
    col = jj >> 8; ic = jj & 255;
  } else {                                           // left + right columns
    const int jj = j - 66560;
    row = 1 + (jj >> 9);
    const int k = jj & 511;
    col = (k < 256) ? 0 : 129; ic = k & 255;
  }
  et1[(((size_t)b * 34 + row) * 130 + col) * 256 + ic] = 0;
}

// ---------------------------------------------------------------------------
// K2: Et[b,oc,p] = sum_c ua_w[oc,c]*enc[b,c,p] + ua_b + uf_b + h1 + att*uf_w
// bf16 WMMA.  grid = (64,4,16), block = 256 (8 waves).
// Epilogue packs 8 consecutive oc per lane -> one b128 store (NHWC+halo).
// ---------------------------------------------------------------------------
__global__ void __launch_bounds__(256) k2_et_gemm(
    const float* __restrict__ enc, const float* __restrict__ ua_w,
    const float* __restrict__ ua_b, const float* __restrict__ uf_w,
    const float* __restrict__ uf_b, const float* __restrict__ h1,
    const float* __restrict__ att, unsigned short* __restrict__ et1)
{
  __shared__ unsigned short lA[64 * 32];   // [oc][k]
  __shared__ unsigned short lB[64 * 32];   // [p][k]  (staged transposed)
  const int tid  = threadIdx.x;
  const int lane = tid & 31;
  const int wave = tid >> 5;
  const int p0   = blockIdx.x * 64;
  const int oc0  = blockIdx.y * 64;
  const int b    = blockIdx.z;

  const int ocsub = wave & 3;
  const int psub0 = (wave >> 2) * 2;

  f32x8 acc0 = {}; f32x8 acc1 = {};

  const int arow  = ocsub * 16 + (lane & 15);
  const int a_klo = (lane < 16) ? 0 : 8;      // ISA A-frag K split
  const int b_klo = (lane < 16) ? 0 : 8;      // ISA B-frag K split

  for (int kb = 0; kb < 32; ++kb) {
    const int k0 = kb << 5;
    __syncthreads();
    #pragma unroll
    for (int it = 0; it < 2; ++it) {                           // stage A: b128 loads
      const int idx = tid + it * 256;
      const int row = idx >> 3, q = idx & 7;
      const float4 v = *reinterpret_cast<const float4*>(
          &ua_w[(size_t)(oc0 + row) * 1024 + k0 + q * 4]);
      uint2 u;
      u.x = (unsigned)f2bf(v.x) | ((unsigned)f2bf(v.y) << 16);
      u.y = (unsigned)f2bf(v.z) | ((unsigned)f2bf(v.w) << 16);
      *reinterpret_cast<uint2*>(&lA[row * 32 + q * 4]) = u;
    }
    #pragma unroll
    for (int it = 0; it < 2; ++it) {                           // stage B^T: b128 loads
      const int idx = tid + it * 256;
      const int k = idx >> 4, q = idx & 15;
      const float4 v = *reinterpret_cast<const float4*>(
          &enc[((size_t)b * 1024 + k0 + k) * 4096 + p0 + q * 4]);
      const int pb = q * 4;
      lB[(pb + 0) * 32 + k] = f2bf(v.x);
      lB[(pb + 1) * 32 + k] = f2bf(v.y);
      lB[(pb + 2) * 32 + k] = f2bf(v.z);
      lB[(pb + 3) * 32 + k] = f2bf(v.w);
    }
    if (kb + 1 < 32)                                           // global_prefetch_b8
      __builtin_prefetch(&enc[((size_t)b * 1024 + k0 + 32 + (tid >> 6)) * 4096 + p0 + (tid & 63)], 0, 1);
    __syncthreads();

    const bf16x16 af = mk_frag(&lA[arow * 32 + a_klo], &lA[arow * 32 + a_klo + 16]);
    const int col0 = ((psub0)     * 16 + (lane & 15)) * 32 + b_klo;
    const int col1 = ((psub0 + 1) * 16 + (lane & 15)) * 32 + b_klo;
    const bf16x16 bf0 = mk_frag(&lB[col0], &lB[col0 + 16]);
    const bf16x16 bf1 = mk_frag(&lB[col1], &lB[col1 + 16]);
    acc0 = __builtin_amdgcn_wmma_f32_16x16x32_bf16(false, af, false, bf0, (short)0, acc0, false, false);
    acc1 = __builtin_amdgcn_wmma_f32_16x16x32_bf16(false, af, false, bf1, (short)0, acc1, false, false);
  }

  // epilogue: lane holds 8 consecutive oc (M=vg+mbase) at one pixel -> b128 store
  const int mbase = (lane < 16) ? 0 : 8;
  const int ocb = oc0 + ocsub * 16 + mbase;
  #pragma unroll
  for (int ps = 0; ps < 2; ++ps) {
    const f32x8 a = ps ? acc1 : acc0;
    const int p = p0 + (psub0 + ps) * 16 + (lane & 15);
    const int h = p >> 7, w = p & 127;
    const float attv = att[b * 4096 + p];
    unsigned short us[8];
    #pragma unroll
    for (int e = 0; e < 8; ++e) {
      const int oce = ocb + e;
      us[e] = f2bf(a[e] + ua_b[oce] + uf_b[oce] + h1[b * 256 + oce] + attv * uf_w[oce]);
    }
    uint4 u;
    u.x = (unsigned)us[0] | ((unsigned)us[1] << 16);
    u.y = (unsigned)us[2] | ((unsigned)us[3] << 16);
    u.z = (unsigned)us[4] | ((unsigned)us[5] << 16);
    u.w = (unsigned)us[6] | ((unsigned)us[7] << 16);
    *reinterpret_cast<uint4*>(
        &et1[(((size_t)b * 34 + h + 1) * 130 + (w + 1)) * 256 + ocb]) = u;
  }
}

// ---------------------------------------------------------------------------
// K3: 3x3 conv 256->256 (implicit GEMM, 9 shifted WMMA GEMMs) fused with
// mask, batchnorm, tanh, dot with v_w, exp*mask.
// grid = (64,1,16), block = 256.  Patch + weight tiles staged via TDM
// (tensor_load_to_lds + s_wait_tensorcnt) when available.
// ---------------------------------------------------------------------------
__global__ void __launch_bounds__(256) k3_conv_attn(
    const unsigned short* __restrict__ et1, const unsigned short* __restrict__ cwbf,
    const float* __restrict__ cb, const float* __restrict__ gamma,
    const float* __restrict__ beta, const float* __restrict__ mean,
    const float* __restrict__ var, const float* __restrict__ vw,
    const float* __restrict__ vb, const int* __restrict__ h_mask,
    const int* __restrict__ w_mask, float* __restrict__ et_exp)
{
  __shared__ unsigned short patch[3 * 66 * 32];  // [row][col][ic]  (ic contiguous)
  __shared__ unsigned short wt[256 * 32];        // [oc][ic]
  __shared__ float etsum[64];

  const int tid = threadIdx.x, lane = tid & 31, wave = tid >> 5;
  const int p0 = blockIdx.x * 64;
  const int b  = blockIdx.z;
  const int h  = p0 >> 7;
  const int w0 = p0 & 127;

  f32x8 acc[2][4];
  #pragma unroll
  for (int i = 0; i < 2; ++i)
    #pragma unroll
    for (int j = 0; j < 4; ++j) acc[i][j] = f32x8{};

  const int a_klo = (lane < 16) ? 0 : 8;
  const int b_klo = (lane < 16) ? 0 : 8;

  for (int icb = 0; icb < 8; ++icb) {
    const int ic0 = icb << 5;
    __syncthreads();
    // ---- stage halo patch: [3 rows][66 cols][32 ic], pure bf16 copy ----
    const unsigned short* gsrc =
        &et1[(((size_t)b * 34 + h) * 130 + w0) * 256 + ic0];
#ifdef HAVE_TDM
    if (tid < 32) {
      // x=ic(32, contig), y=col(66, stride 256), z=row(3, stride 130*256)
      tdm_load_bf16(gsrc, patch, 256u, 130u, 34u, 32u, 66u, 3u, 256u, 33280u);
      __builtin_amdgcn_s_wait_tensorcnt(0);
    }
#else
    {
      const unsigned* s32 = reinterpret_cast<const unsigned*>(gsrc);
      unsigned* d32 = reinterpret_cast<unsigned*>(patch);
      for (int idx = tid; idx < 3 * 66 * 16; idx += 256) {
        const int q = idx & 15;          // uint within the 32-ic run
        const int rc = idx >> 4;
        const int row = rc / 66, cc = rc - row * 66;
        d32[idx] = s32[((size_t)row * 130 * 128) + (size_t)cc * 128 + q];
      }
    }
#endif
    for (int t = 0; t < 9; ++t) {
      __syncthreads();
      // ---- stage weight slab: [256 oc][32 ic] from cwbf[t][oc][ic0..] ----
#ifdef HAVE_TDM
      if (tid < 32) {
        // x=ic(32, contig), y=oc(256, stride 256)
        tdm_load_bf16(&cwbf[(size_t)t * 65536 + ic0], wt,
                      256u, 256u, 0u, 32u, 256u, 0u, 256u, 0u);
        __builtin_amdgcn_s_wait_tensorcnt(0);
      }
#else
      {
        const unsigned* s32 = reinterpret_cast<const unsigned*>(&cwbf[(size_t)t * 65536 + ic0]);
        unsigned* d32 = reinterpret_cast<unsigned*>(wt);
        for (int idx = tid; idx < 4096; idx += 256) {
          const int oc = idx >> 4, q = idx & 15;
          d32[idx] = s32[(size_t)oc * 128 + q];
        }
      }
#endif
      __syncthreads();
      const int row = t / 3, dw = t - row * 3;
      bf16x16 bfr[4];
      #pragma unroll
      for (int ns = 0; ns < 4; ++ns) {
        const int cc = ns * 16 + (lane & 15) + dw;
        const unsigned short* pp = &patch[(row * 66 + cc) * 32 + b_klo];
        bfr[ns] = mk_frag(pp, pp + 16);
      }
      #pragma unroll
      for (int mi = 0; mi < 2; ++mi) {
        const int orow = (wave * 2 + mi) * 16 + (lane & 15);
        const bf16x16 af = mk_frag(&wt[orow * 32 + a_klo], &wt[orow * 32 + a_klo + 16]);
        #pragma unroll
        for (int ns = 0; ns < 4; ++ns)
          acc[mi][ns] = __builtin_amdgcn_wmma_f32_16x16x32_bf16(
              false, af, false, bfr[ns], (short)0, acc[mi][ns], false, false);
      }
    }
  }

  __syncthreads();
  if (tid < 64) etsum[tid] = 0.f;
  __syncthreads();

  const int hm = h_mask[b], wm = w_mask[b];
  const float hmask = (h < hm) ? 1.f : 0.f;
  const int mbase = (lane < 16) ? 0 : 8;
  for (int mi = 0; mi < 2; ++mi) {
    #pragma unroll
    for (int vg = 0; vg < 8; ++vg) {
      const int oc = (wave * 2 + mi) * 16 + vg + mbase;
      const float scale = gamma[oc] * rsqrtf(var[oc] + 1e-5f);
      const float shift = beta[oc] - mean[oc] * scale;
      const float vwoc  = vw[oc];
      const float bias  = cb[oc];
      #pragma unroll
      for (int ns = 0; ns < 4; ++ns) {
        const int n = ns * 16 + (lane & 15);
        const float m = ((w0 + n) < wm) ? hmask : 0.f;
        float v = (acc[mi][ns][vg] + bias) * m;      // mask before BN, per reference
        v = tanhf(v * scale + shift);
        atomicAdd(&etsum[n], v * vwoc);              // ds_add_f32 over channels
      }
    }
  }
  __syncthreads();
  if (tid < 64) {
    const int p = p0 + tid;
    const float m = (h < hm && (w0 + tid) < wm) ? 1.f : 0.f;
    et_exp[b * 4096 + p] = expf(etsum[tid] + vb[0]) * m;
  }
}

// ---------------------------------------------------------------------------
__global__ void __launch_bounds__(256) k4_reduce(
    const float* __restrict__ et_exp, float* __restrict__ etsum_b)
{
  __shared__ float red[256];
  const int b = blockIdx.x, tid = threadIdx.x;
  float s = 0.f;
  for (int p = tid; p < 4096; p += 256) s += et_exp[b * 4096 + p];
  red[tid] = s; __syncthreads();
  for (int off = 128; off > 0; off >>= 1) {
    if (tid < off) red[tid] += red[tid + off];
    __syncthreads();
  }
  if (tid == 0) etsum_b[b] = red[0];
}

__global__ void __launch_bounds__(256) k5_ct(
    const float* __restrict__ et_exp, const float* __restrict__ etsum_b,
    const float* __restrict__ enc, float* __restrict__ ct)
{
  const int b = blockIdx.y;
  const int c = blockIdx.x * 8 + (threadIdx.x >> 5);
  const int lane = threadIdx.x & 31;
  const float* e = &enc[((size_t)b * 1024 + c) * 4096];
  const float* w = &et_exp[b * 4096];
  float s = 0.f;
  for (int p = lane; p < 4096; p += 32) s += e[p] * w[p];
  for (int off = 16; off > 0; off >>= 1) s += __shfl_xor(s, off, 32);
  if (lane == 0) ct[b * 1024 + c] = s / (etsum_b[b] + 1e-8f);
}

__global__ void __launch_bounds__(256) k5b_div(
    const float* __restrict__ et_exp, const float* __restrict__ etsum_b,
    float* __restrict__ out2)
{
  const int i = blockIdx.x * 256 + threadIdx.x;
  out2[i] = et_exp[i] / (etsum_b[i >> 12] + 1e-8f);
}

// ---------------------------------------------------------------------------
__global__ void __launch_bounds__(256) k6_final(
    const float* __restrict__ ws_st, const float* __restrict__ ws_emb,
    const float* __restrict__ ct,
    const float* __restrict__ wih, const float* __restrict__ whh,
    const float* __restrict__ bih, const float* __restrict__ bhh,
    const float* __restrict__ h2w, const float* __restrict__ h2b,
    const float* __restrict__ e2w, const float* __restrict__ e2b,
    const float* __restrict__ wcw, const float* __restrict__ wcb,
    const float* __restrict__ ow,  const float* __restrict__ ob,
    float* __restrict__ out0, float* __restrict__ out1)
{
  __shared__ float lct[1024], lst[256], lhn[256], lemb[256], lpre[128], llog[512], lred[2];
  const int b = blockIdx.x, t = threadIdx.x;
  for (int i = t; i < 1024; i += 256) lct[i] = ct[b * 1024 + i];
  lst[t]  = ws_st[b * 256 + t];
  lemb[t] = ws_emb[b * 256 + t];
  __syncthreads();
  float gr = 0.f, gz = 0.f, gn = 0.f;
  for (int k = 0; k < 1024; ++k) {
    const float x = lct[k];
    gr += x * wih[(size_t)(t)       * 1024 + k];
    gz += x * wih[(size_t)(256 + t) * 1024 + k];
    gn += x * wih[(size_t)(512 + t) * 1024 + k];
  }
  float hr = 0.f, hz = 0.f, hn = 0.f;
  for (int k = 0; k < 256; ++k) {
    const float x = lst[k];
    hr += x * whh[(t)       * 256 + k];
    hz += x * whh[(256 + t) * 256 + k];
    hn += x * whh[(512 + t) * 256 + k];
  }
  const float r = sigmoidf(gr + bih[t]       + hr + bhh[t]);
  const float z = sigmoidf(gz + bih[256 + t] + hz + bhh[256 + t]);
  const float n = tanhf(gn + bih[512 + t] + r * (hn + bhh[512 + t]));
  const float hnext = (1.f - z) * n + z * lst[t];
  lhn[t] = hnext;
  out1[b * 256 + t] = hnext;
  __syncthreads();
  if (t < 128) {
    float acc = h2b[t] + e2b[t] + wcb[t];
    for (int k = 0; k < 256; ++k) acc += lhn[k] * h2w[t * 256 + k] + lemb[k] * e2w[t * 256 + k];
    for (int k = 0; k < 1024; ++k) acc += lct[k] * wcw[(size_t)t * 1024 + k];
    lpre[t] = acc;
  }
  __syncthreads();
  #pragma unroll
  for (int vv = 0; vv < 2; ++vv) {
    const int v = t + vv * 256;
    float acc = ob[v];
    for (int k = 0; k < 128; ++k) acc += lpre[k] * ow[v * 128 + k];
    llog[v] = acc;
  }
  __syncthreads();
  if (t == 0) {
    float mx = llog[0];
    for (int v = 1; v < 512; ++v) mx = fmaxf(mx, llog[v]);
    float se = 0.f;
    for (int v = 0; v < 512; ++v) se += expf(llog[v] - mx);
    lred[0] = mx; lred[1] = logf(se);
  }
  __syncthreads();
  const float lse = lred[0] + lred[1];
  #pragma unroll
  for (int vv = 0; vv < 2; ++vv) {
    const int v = t + vv * 256;
    out0[b * 512 + v] = llog[v] - lse;
  }
}

// ---------------------------------------------------------------------------
extern "C" void kernel_launch(void* const* d_in, const int* in_sizes, int n_in,
                              void* d_out, int out_size, void* d_ws, size_t ws_size,
                              hipStream_t stream) {
  (void)in_sizes; (void)n_in; (void)out_size; (void)ws_size;
  const int*   input_a  = (const int*)  d_in[0];
  const float* hidden   = (const float*)d_in[1];
  const float* enc      = (const float*)d_in[2];
  const float* att_in   = (const float*)d_in[3];
  const float* dec_att  = (const float*)d_in[4];
  const int*   h_mask   = (const int*)  d_in[5];
  const int*   w_mask   = (const int*)  d_in[6];
  const float* emb_table = (const float*)d_in[10];
  const float* gru1_wih  = (const float*)d_in[11];
  const float* gru1_whh  = (const float*)d_in[12];
  const float* gru1_bih  = (const float*)d_in[13];
  const float* gru1_bhh  = (const float*)d_in[14];
  const float* gru_wih   = (const float*)d_in[15];
  const float* gru_whh   = (const float*)d_in[16];
  const float* gru_bih   = (const float*)d_in[17];
  const float* gru_bhh   = (const float*)d_in[18];
  const float* hidden_w  = (const float*)d_in[19];
  const float* hidden_b  = (const float*)d_in[20];
  const float* ua_w      = (const float*)d_in[21];
  const float* ua_b      = (const float*)d_in[22];
  const float* uf_w      = (const float*)d_in[23];
  const float* uf_b      = (const float*)d_in[24];
  const float* v_w       = (const float*)d_in[25];
  const float* v_b       = (const float*)d_in[26];
  const float* conv1_w   = (const float*)d_in[27];
  const float* conv1_b   = (const float*)d_in[28];
  const float* convtan_w = (const float*)d_in[29];
  const float* convtan_b = (const float*)d_in[30];
  const float* bn1_gamma = (const float*)d_in[31];
  const float* bn1_beta  = (const float*)d_in[32];
  const float* bn1_mean  = (const float*)d_in[33];
  const float* bn1_var   = (const float*)d_in[34];
  const float* hidden2_w = (const float*)d_in[35];
  const float* hidden2_b = (const float*)d_in[36];
  const float* emb2_w    = (const float*)d_in[37];
  const float* emb2_b    = (const float*)d_in[38];
  const float* wc_w      = (const float*)d_in[39];
  const float* wc_b      = (const float*)d_in[40];
  const float* out_w     = (const float*)d_in[41];
  const float* out_b     = (const float*)d_in[42];

  // outputs: output[16,1,512] | hidden[16,1,256] | et_div[16,1,32,128] | att_sum[16,1,32,128]
  float* out0 = (float*)d_out;
  float* out1 = out0 + 8192;
  float* out2 = out1 + 4096;
  float* out3 = out2 + 65536;

  // workspace (floats)
  float* wsf      = (float*)d_ws;
  float* ws_emb   = wsf;                   // 4096
  float* ws_st    = wsf + 4096;            // 4096
  float* ws_h1    = wsf + 8192;            // 4096
  float* ws_ct    = wsf + 12288;           // 16384
  float* ws_etexp = wsf + 28672;           // 65536
  float* ws_etsum = wsf + 94208;           // 16 (+ pad)
  unsigned short* ws_cwbf = (unsigned short*)(wsf + 94224);   // 589824 bf16
  unsigned short* ws_et1  = (unsigned short*)(wsf + 389136);  // [16][34][130][256] bf16

  k0_prep<<<16, 256, 0, stream>>>(input_a, hidden, emb_table,
                                  gru1_wih, gru1_whh, gru1_bih, gru1_bhh,
                                  hidden_w, hidden_b, ws_emb, ws_st, ws_h1);
  k1_att<<<256, 256, 0, stream>>>(att_in, dec_att, conv1_w, conv1_b, out3);
  k_wprep<<<2304, 256, 0, stream>>>(convtan_w, ws_cwbf);
  k_zero_halo<<<5184, 256, 0, stream>>>(ws_et1);
  k2_et_gemm<<<dim3(64, 4, 16), 256, 0, stream>>>(enc, ua_w, ua_b, uf_w, uf_b,
                                                  ws_h1, out3, ws_et1);
  k3_conv_attn<<<dim3(64, 1, 16), 256, 0, stream>>>(ws_et1, ws_cwbf, convtan_b,
                                                    bn1_gamma, bn1_beta, bn1_mean, bn1_var,
                                                    v_w, v_b, h_mask, w_mask, ws_etexp);
  k4_reduce<<<16, 256, 0, stream>>>(ws_etexp, ws_etsum);
  k5_ct<<<dim3(128, 16), 256, 0, stream>>>(ws_etexp, ws_etsum, enc, ws_ct);
  k5b_div<<<256, 256, 0, stream>>>(ws_etexp, ws_etsum, out2);
  k6_final<<<16, 256, 0, stream>>>(ws_st, ws_emb, ws_ct,
                                   gru_wih, gru_whh, gru_bih, gru_bhh,
                                   hidden2_w, hidden2_b, emb2_w, emb2_b,
                                   wc_w, wc_b, out_w, out_b, out0, out1);
}
